// FullGraphEncoder_19129784336896
// MI455X (gfx1250) — matrix operand order repaired and checked by reading, not compiled, not run on hardware
//
#include <hip/hip_runtime.h>
#include <hip/hip_bf16.h>
#include <math.h>

// ---------------- problem constants ----------------
#define VOCAB   100
#define NBOND   4
#define EMB     128
#define DEPTH   4
#define NNODES  25000
#define NEDGES  50000
#define MAXNB   6

// nodes padded to multiple of 16 for WMMA M tiles (25008 = 1563*16)
#define MPAD_NODES 25008
#define MTILES_E   (NEDGES / 16)      // 3125 (exact)
#define MTILES_N   (MPAD_NODES / 16)  // 1563

typedef _Float16 v16h __attribute__((ext_vector_type(16)));
typedef _Float16 v8h  __attribute__((ext_vector_type(8)));
typedef float    v8f  __attribute__((ext_vector_type(8)));

// ---------------------------------------------------------------------------
// Pack weight matrix W_h (the bottom 128 rows of a [din,128] row-major weight)
// into WMMA B-operand per-lane layout:
//   index = ((ntile*4 + kstep)*32 + lane)*16 + j
//   B element = W[(row0 + kstep*32 + (lane>>4)*16 + j)*128 + col(ntile,lane)]
// For the 4-way concat (N=512) the column selects among Wi/Wo/Wu/Wf.
// ---------------------------------------------------------------------------
__global__ __launch_bounds__(256) void pack_b4_kernel(
    const float* __restrict__ Wi, const float* __restrict__ Wo,
    const float* __restrict__ Wu, const float* __restrict__ Wf,
    _Float16* __restrict__ Bp)
{
    int t = blockIdx.x * 256 + threadIdx.x;          // 32 ntiles * 4 ksteps * 32 lanes
    if (t >= 32 * 4 * 32) return;
    int lane  = t & 31;
    int s     = (t >> 5) & 3;
    int ntile = t >> 7;
    int n     = ntile * 16 + (lane & 15);            // 0..511
    int w     = n >> 7;                              // which weight matrix
    int col   = n & 127;
    int kb    = (lane >> 4) << 4;                    // 0 or 16
    const float* W = (w == 0) ? Wi : (w == 1) ? Wo : (w == 2) ? Wu : Wf;
    _Float16* dst = Bp + (size_t)t * 16;
#pragma unroll
    for (int j = 0; j < 16; ++j) {
        int K = s * 32 + kb + j;                     // 0..127 within W_h
        dst[j] = (_Float16)W[(size_t)(VOCAB + NBOND + K) * EMB + col];
    }
}

__global__ __launch_bounds__(256) void pack_bt_kernel(
    const float* __restrict__ Wt, _Float16* __restrict__ Bpt)
{
    int t = blockIdx.x * 256 + threadIdx.x;          // 8 ntiles * 4 ksteps * 32 lanes
    if (t >= 8 * 4 * 32) return;
    int lane  = t & 31;
    int s     = (t >> 5) & 3;
    int ntile = t >> 7;
    int col   = ntile * 16 + (lane & 15);            // 0..127
    int kb    = (lane >> 4) << 4;
    _Float16* dst = Bpt + (size_t)t * 16;
#pragma unroll
    for (int j = 0; j < 16; ++j) {
        int K = s * 32 + kb + j;
        dst[j] = (_Float16)Wt[(size_t)(VOCAB + K) * EMB + col];
    }
}

// Zero initial LSTM state: h (f16) and c (f32)
__global__ __launch_bounds__(256) void zero_state_kernel(
    _Float16* __restrict__ hF, float* __restrict__ c0)
{
    int t = blockIdx.x * 256 + threadIdx.x;
    if (t >= NEDGES * EMB) return;
    hF[t] = (_Float16)0.0f;
    c0[t] = 0.0f;
}

// ---------------------------------------------------------------------------
// WMMA GEMM: C[M x N] = A[Mtiles*16 x 128](f16,row-major) * Bpacked[128 x N]
// N in {128, 512}. Block = 256 threads = 8 waves; each wave owns a 16x64
// output tile (4 accumulators). The block's A m-tiles (8/nquads of them, each
// a contiguous 4KB row-block) are staged cooperatively in LDS and read back
// as per-lane WMMA fragments via ds_load_b128. B fragments are software-
// pipelined (register double-buffer) so step s+1 loads overlap step-s WMMAs.
// 16 x v_wmma_f32_16x16x32_f16 per wave.
// ---------------------------------------------------------------------------
__global__ __launch_bounds__(256) void wmma_gemm_kernel(
    const _Float16* __restrict__ A, const _Float16* __restrict__ Bp,
    float* __restrict__ C, int Mtiles, int N)
{
    __shared__ _Float16 As[4 * 16 * 128];            // up to 4 staged m-tiles (16KB)
    const int lane   = threadIdx.x & 31;
    const int wave   = threadIdx.x >> 5;
    const int nquads = N >> 6;                       // 8 (N=512) or 2 (N=128)
    const int mq     = 8 / nquads;                   // m-tiles per block: 1 or 4

    // ---- cooperative stage of the block's A rows (contiguous mq*2048 halfs) ----
    const size_t blockBase  = (size_t)blockIdx.x * mq * 2048;
    const size_t totalHalfs = (size_t)Mtiles * 2048;
    for (int i = 0; i < mq; ++i) {
        size_t idx = blockBase + (size_t)i * 2048 + (size_t)threadIdx.x * 8;
        v8h v;
        if (idx < totalHalfs) {
            v = *(const v8h*)(A + idx);
        } else {
#pragma unroll
            for (int j = 0; j < 8; ++j) v[j] = (_Float16)0.0f;
        }
        *(v8h*)(&As[i * 2048 + threadIdx.x * 8]) = v;
    }
    __syncthreads();

    const int tile = blockIdx.x * 8 + wave;
    if (tile >= Mtiles * nquads) return;             // safe: post-barrier
    const int mtile = tile / nquads;
    const int nquad = tile % nquads;
    const int mloc  = wave / nquads;                 // staged m-tile index in LDS
    const _Float16* as = &As[mloc * 2048];

    const int m  = lane & 15;
    const int hi = lane >> 4;                        // lane group
    const int kb = hi << 4;                          // A: 0 or 16

    v8f acc[4];
#pragma unroll
    for (int q = 0; q < 4; ++q)
#pragma unroll
        for (int r = 0; r < 8; ++r) acc[q][r] = 0.0f;

    // ---- software-pipelined k loop: B double-buffered in registers ----
    v16h bCur[4], bNxt[4];
#pragma unroll
    for (int q = 0; q < 4; ++q)
        bCur[q] = *((const v16h*)Bp + (size_t)(((nquad * 4 + q) * 4 + 0) * 32 + lane));

#pragma unroll
    for (int s = 0; s < 4; ++s) {
        if (s < 3) {
#pragma unroll
            for (int q = 0; q < 4; ++q)
                bNxt[q] = *((const v16h*)Bp +
                            (size_t)(((nquad * 4 + q) * 4 + (s + 1)) * 32 + lane));
        }
        const int k0 = s * 32;
        // A fragment from LDS: lanes 0-15 cover K {k0+0..15};
        //                      lanes 16-31 cover K {k0+16..31}
        v8h lo = *(const v8h*)(as + m * 128 + k0 + kb);
        v8h up = *(const v8h*)(as + m * 128 + k0 + kb + 8);
        v16h a;
#pragma unroll
        for (int j = 0; j < 8; ++j) { a[j] = lo[j]; a[j + 8] = up[j]; }
#pragma unroll
        for (int q = 0; q < 4; ++q) {
            acc[q] = __builtin_amdgcn_wmma_f32_16x16x32_f16(
                false, a, false, bCur[q], (short)0, acc[q], false, false);
        }
        if (s < 3) {
#pragma unroll
            for (int q = 0; q < 4; ++q) bCur[q] = bNxt[q];
        }
    }

    // C/D layout: VGPR r -> row (r + 8*hi), col = lane&15 within the 16x16 tile
    const int row0 = mtile * 16 + 8 * hi;
#pragma unroll
    for (int q = 0; q < 4; ++q) {
        float* cc = C + (size_t)row0 * N + nquad * 64 + q * 16 + (lane & 15);
#pragma unroll
        for (int r = 0; r < 8; ++r) cc[(size_t)r * N] = acc[q][r];
    }
}

// ---------------------------------------------------------------------------
// Fused LSTM step: gathers hW projections over bond_list, builds the one-hot
// bias rows on the fly (fmess @ W == W[row(nid)] + W[100+bond] + b), applies
// gates, edge-0 mask; writes new h (f16 for next WMMA GEMM) and c (ping-pong).
// ---------------------------------------------------------------------------
__device__ __forceinline__ float sigf(float x) { return 1.0f / (1.0f + __expf(-x)); }

__global__ __launch_bounds__(256) void lstm_step_kernel(
    const float* __restrict__ hW,        // [E,512] = [i|o|u|f] projections of h
    const float* __restrict__ c_in,      // [E,128]
    const int*   __restrict__ bond,      // [E,6]
    const int*   __restrict__ node_ids, const int* __restrict__ edge_src,
    const int*   __restrict__ edge_type,
    const float* __restrict__ Wi, const float* __restrict__ bi,
    const float* __restrict__ Wo, const float* __restrict__ bo,
    const float* __restrict__ Wu, const float* __restrict__ bu,
    const float* __restrict__ Wf, const float* __restrict__ bf,
    _Float16* __restrict__ hF,           // out [E,128] f16
    float* __restrict__ c_out)           // out [E,128]
{
    int t = blockIdx.x * 256 + threadIdx.x;
    if (t >= NEDGES * EMB) return;
    const int e = t >> 7, j = t & 127;

    const int nid = node_ids[edge_src[e]];
    const int bty = edge_type[e];
    const size_t r0 = (size_t)nid * EMB + j;
    const size_t r1 = (size_t)(VOCAB + bty) * EMB + j;
    const float pi = bi[j] + Wi[r0] + Wi[r1];
    const float po = bo[j] + Wo[r0] + Wo[r1];
    const float pu = bu[j] + Wu[r0] + Wu[r1];
    const float pf = bf[j] + Wf[r0] + Wf[r1];

    float si = 0.f, so = 0.f, su = 0.f, fc = 0.f;
#pragma unroll
    for (int k = 0; k < MAXNB; ++k) {
        const int b = bond[e * MAXNB + k];
        const float* hwb = hW + (size_t)b * 512 + j;
        si += hwb[0];
        so += hwb[128];
        su += hwb[256];
        const float fk = sigf(hwb[384] + pf);
        fc += fk * c_in[(size_t)b * EMB + j];
    }
    const float ig = sigf(si + pi);
    const float og = sigf(so + po);
    const float ug = tanhf(su + pu);
    float cn = ig * ug + fc;
    const float msk = (e == 0) ? 0.0f : 1.0f;   // emask: padding edge 0
    cn *= msk;
    c_out[t] = cn;
    hF[t] = (_Float16)(og * tanhf(cn) * msk);
}

// Neighbor message gather-sum over adj_list -> f16 A-operand (padded rows = 0)
__global__ __launch_bounds__(256) void nei_gather_kernel(
    const _Float16* __restrict__ hF, const int* __restrict__ adj,
    _Float16* __restrict__ nf)
{
    int t = blockIdx.x * 256 + threadIdx.x;
    if (t >= MPAD_NODES * EMB) return;
    const int n = t >> 7, j = t & 127;
    float s = 0.0f;
    if (n < NNODES) {
#pragma unroll
        for (int k = 0; k < MAXNB; ++k)
            s += (float)hF[(size_t)adj[n * MAXNB + k] * EMB + j];
    }
    nf[t] = (_Float16)s;
}

// Epilogue: out = relu(gemm + bt + Wt[node_id-row]) with node-0 mask
__global__ __launch_bounds__(256) void out_epilogue_kernel(
    const float* __restrict__ tmp, const int* __restrict__ node_ids,
    const float* __restrict__ Wt, const float* __restrict__ bt,
    float* __restrict__ out)
{
    int t = blockIdx.x * 256 + threadIdx.x;
    if (t >= NNODES * EMB) return;
    const int n = t >> 7, j = t & 127;
    float v = tmp[(size_t)n * EMB + j] + bt[j] + Wt[(size_t)node_ids[n] * EMB + j];
    v = fmaxf(v, 0.0f);
    out[t] = (n == 0) ? 0.0f : v;
}

// ---------------------------------------------------------------------------
extern "C" void kernel_launch(void* const* d_in, const int* in_sizes, int n_in,
                              void* d_out, int out_size, void* d_ws, size_t ws_size,
                              hipStream_t stream)
{
    const int*   node_ids  = (const int*)  d_in[0];
    const int*   edge_src  = (const int*)  d_in[1];
    const int*   edge_type = (const int*)  d_in[2];
    const int*   adj_list  = (const int*)  d_in[3];
    const int*   bond_list = (const int*)  d_in[4];
    const float* Wi = (const float*)d_in[5];
    const float* bi = (const float*)d_in[6];
    const float* Wo = (const float*)d_in[7];
    const float* bo = (const float*)d_in[8];
    const float* Wf = (const float*)d_in[9];
    const float* bf = (const float*)d_in[10];
    const float* Wu = (const float*)d_in[11];
    const float* bu = (const float*)d_in[12];
    const float* Wt = (const float*)d_in[13];
    const float* bt = (const float*)d_in[14];
    float* out = (float*)d_out;

    // ---- workspace carve-up (all offsets 256B-aligned; ~186 MB total) ----
    char* ws = (char*)d_ws;
    size_t off = 0;
    _Float16* Bp  = (_Float16*)(ws + off); off += (size_t)32 * 4 * 32 * 16 * 2;     // 131072
    _Float16* Bpt = (_Float16*)(ws + off); off += (size_t)8  * 4 * 32 * 16 * 2;     // 32768
    _Float16* hF  = (_Float16*)(ws + off); off += (size_t)NEDGES * EMB * 2;         // 12.8 MB
    float*    c0  = (float*)   (ws + off); off += (size_t)NEDGES * EMB * 4;         // 25.6 MB
    float*    c1  = (float*)   (ws + off); off += (size_t)NEDGES * EMB * 4;         // 25.6 MB
    float*    hW  = (float*)   (ws + off); off += (size_t)NEDGES * 512 * 4;         // 102.4 MB
    _Float16* nf  = (_Float16*)(ws + off); off += (size_t)MPAD_NODES * EMB * 2;     // 6.4 MB
    float*    tmp = (float*)   (ws + off); off += (size_t)MPAD_NODES * EMB * 4;     // 12.8 MB
    (void)ws_size; (void)in_sizes; (void)n_in; (void)out_size;

    // ---- pack weights into WMMA B layout + zero initial state ----
    pack_b4_kernel<<<(32 * 4 * 32 + 255) / 256, 256, 0, stream>>>(Wi, Wo, Wu, Wf, Bp);
    pack_bt_kernel<<<(8 * 4 * 32 + 255) / 256, 256, 0, stream>>>(Wt, Bpt);
    zero_state_kernel<<<(NEDGES * EMB + 255) / 256, 256, 0, stream>>>(hF, c0);

    // ---- DEPTH iterations: one fused GEMM + one fused gather/LSTM step ----
    float* cin = c0; float* cout = c1;
    for (int d = 0; d < DEPTH; ++d) {
        // hW[E,512] = h @ [Wi_h | Wo_h | Wu_h | Wf_h]
        wmma_gemm_kernel<<<(MTILES_E * 8 + 7) / 8, 256, 0, stream>>>(hF, Bp, hW, MTILES_E, 512);
        lstm_step_kernel<<<(NEDGES * EMB + 255) / 256, 256, 0, stream>>>(
            hW, cin, bond_list, node_ids, edge_src, edge_type,
            Wi, bi, Wo, bo, Wu, bu, Wf, bf, hF, cout);
        float* t2 = cin; cin = cout; cout = t2;
    }

    // ---- readout: neighbor sum, GEMM vs Wt_h, fused epilogue ----
    nei_gather_kernel<<<(MPAD_NODES * EMB + 255) / 256, 256, 0, stream>>>(hF, adj_list, nf);
    wmma_gemm_kernel<<<(MTILES_N * 2 + 7) / 8, 256, 0, stream>>>(nf, Bpt, tmp, MTILES_N, 128);
    out_epilogue_kernel<<<(NNODES * EMB + 255) / 256, 256, 0, stream>>>(tmp, node_ids, Wt, bt, out);
}
